// PathDistance_61598420959480
// MI455X (gfx1250) — matrix operand order repaired and checked by reading, not compiled, not run on hardware
//
#include <hip/hip_runtime.h>

typedef __attribute__((ext_vector_type(16))) __bf16        v16bf;
typedef __attribute__((ext_vector_type(8)))  float         v8f;
typedef __attribute__((ext_vector_type(4)))  unsigned int  u32v4;
typedef __attribute__((ext_vector_type(4)))  float         f32v4;
typedef unsigned long long u64;

union FragBF { v16bf v; u32v4 u[2]; };

#define DDIM   64    // feature dim
#define KTOP   16    // top-k
#define QPB    32    // queries per block (2 WMMA N-tiles, shared A operand)
#define NT     16    // database rows per WMMA tile (WMMA M)
#define NWAVES 8
#define NSPLIT 2     // database halves (block groups)

__device__ __forceinline__ unsigned short f32_to_bf16_rne(float f) {
  unsigned int u = __float_as_uint(f);
  unsigned int r = u + 0x7FFFu + ((u >> 16) & 1u);   // round-to-nearest-even
  return (unsigned short)(r >> 16);
}
__device__ __forceinline__ float bf16u_to_f32(unsigned short h) {
  return __uint_as_float(((unsigned int)h) << 16);
}

// ---------------------------------------------------------------------------
// Kernel A: split f32 rows into bf16 hi/lo planes + row squared-norms.
// ---------------------------------------------------------------------------
__global__ void knn_convert(const float* __restrict__ x, const float* __restrict__ y,
                            unsigned short* __restrict__ xh, unsigned short* __restrict__ xl,
                            unsigned short* __restrict__ yh, unsigned short* __restrict__ yl,
                            float* __restrict__ x2, float* __restrict__ y2,
                            int B1, int B2) {
  int gid = blockIdx.x * blockDim.x + threadIdx.x;
  const float* src; unsigned int* dh; unsigned int* dl; float* nrm;
  if (gid < B2) {
    src = y + (size_t)gid * DDIM;
    dh  = (unsigned int*)(yh + (size_t)gid * DDIM);
    dl  = (unsigned int*)(yl + (size_t)gid * DDIM);
    nrm = y2 + gid;
  } else if (gid < B2 + B1) {
    int row = gid - B2;
    src = x + (size_t)row * DDIM;
    dh  = (unsigned int*)(xh + (size_t)row * DDIM);
    dl  = (unsigned int*)(xl + (size_t)row * DDIM);
    nrm = x2 + row;
  } else {
    return;
  }
  float acc = 0.f;
#pragma unroll
  for (int j = 0; j < DDIM / 2; ++j) {
    float a = src[2 * j], b = src[2 * j + 1];
    acc = fmaf(a, a, acc);
    acc = fmaf(b, b, acc);
    unsigned short ah = f32_to_bf16_rne(a), bh = f32_to_bf16_rne(b);
    unsigned short al = f32_to_bf16_rne(a - bf16u_to_f32(ah));
    unsigned short bl = f32_to_bf16_rne(b - bf16u_to_f32(bh));
    dh[j] = (unsigned int)ah | ((unsigned int)bh << 16);
    dl[j] = (unsigned int)al | ((unsigned int)bl << 16);
  }
  *nrm = acc;
}

// ---------------------------------------------------------------------------
// Lane-queue compaction: merge 16 appended keys into the sorted keep-16.
// Slots [0..15] = sorted keep, [16..31] = append buffer (reset to UMAX).
// ---------------------------------------------------------------------------
__device__ __forceinline__ void lane_compact(u64* mb, float& worst, int& cnt) {
  u64 a[16], b[16];
#pragma unroll
  for (int i = 0; i < 16; ++i) a[i] = mb[i];
#pragma unroll
  for (int i = 0; i < 16; ++i) b[i] = mb[16 + i];
  // bitonic sort b ascending (80 compare-exchanges)
#pragma unroll
  for (int sz = 2; sz <= 16; sz <<= 1) {
#pragma unroll
    for (int st = sz >> 1; st >= 1; st >>= 1) {
#pragma unroll
      for (int i = 0; i < 16; ++i) {
        int j = i ^ st;
        if (j > i) {
          bool up = ((i & sz) == 0);
          u64 xv = b[i], yv = b[j];
          bool sw = up ? (xv > yv) : (xv < yv);
          b[i] = sw ? yv : xv;
          b[j] = sw ? xv : yv;
        }
      }
    }
  }
  // a asc ++ reverse(b) is bitonic; first merge stage picks the lower 16
  u64 lo[16];
#pragma unroll
  for (int i = 0; i < 16; ++i) {
    u64 xv = a[i], yv = b[15 - i];
    lo[i] = (xv < yv) ? xv : yv;
  }
  // clean the bitonic lower half: strides 8,4,2,1 (32 compare-exchanges)
#pragma unroll
  for (int st = 8; st >= 1; st >>= 1) {
#pragma unroll
    for (int i = 0; i < 16; ++i) {
      int j = i ^ st;
      if (j > i) {
        u64 xv = lo[i], yv = lo[j];
        bool sw = xv > yv;
        lo[i] = sw ? yv : xv;
        lo[j] = sw ? xv : yv;
      }
    }
  }
#pragma unroll
  for (int i = 0; i < 16; ++i) mb[i] = lo[i];
#pragma unroll
  for (int i = 0; i < 16; ++i) mb[16 + i] = ~0ull;
  worst = __uint_as_float((unsigned int)(lo[15] >> 32));
  cnt = 0;
}

// ---------------------------------------------------------------------------
// Kernel B: fused distance GEMM. Each A (y) fragment feeds TWO query tiles
// (12 WMMAs per 10 b128 loads), halving L2 database traffic. Blocks are split
// into 2 groups over database halves; per-(query,half) sorted-16 runs go to
// workspace for the final 2-way merge kernel.
// ---------------------------------------------------------------------------
__global__ __launch_bounds__(256) void knn_main(
    const unsigned short* __restrict__ xh, const unsigned short* __restrict__ xl,
    const unsigned short* __restrict__ yh, const unsigned short* __restrict__ yl,
    const float* __restrict__ x2, const float* __restrict__ y2,
    u64* __restrict__ cand,        // [B1][NSPLIT][KTOP] packed (d2bits<<32)|idx
    int B1, int B2) {
  const int lane    = threadIdx.x & 31;
  const int wave    = threadIdx.x >> 5;
  const int qcol    = lane & 15;   // WMMA N index
  const int hf      = lane >> 4;   // lane half: K group (A/B), M offset (D)
  const int ngroups = B1 / QPB;                 // 128
  const int qbase   = (blockIdx.x % ngroups) * QPB;
  const int nhalf   = blockIdx.x / ngroups;     // database half
  const int qA      = qbase + qcol;
  const int qB      = qbase + 16 + qcol;

  // [wave][lane][queue*32 + slot]: per queue 0-15 sorted keep, 16-31 append
  __shared__ u64 cbuf[NWAVES][32][64];
  __shared__ int heads[QPB][16];
  u64* q0 = &cbuf[wave][lane][0];
  u64* q1 = &cbuf[wave][lane][32];
#pragma unroll
  for (int s = 0; s < 64; ++s) q0[s] = ~0ull;

  // --- B fragments for both query tiles: resident for the whole scan. ---
  FragBF bAh0, bAh1, bAl0, bAl1, bBh0, bBh1, bBl0, bBl1;
  {
    const unsigned short* ph = xh + (size_t)qA * DDIM + hf * 16;
    const unsigned short* pl = xl + (size_t)qA * DDIM + hf * 16;
    bAh0.u[0] = ((const u32v4*)ph)[0];        bAh0.u[1] = ((const u32v4*)ph)[1];
    bAh1.u[0] = ((const u32v4*)(ph + 32))[0]; bAh1.u[1] = ((const u32v4*)(ph + 32))[1];
    bAl0.u[0] = ((const u32v4*)pl)[0];        bAl0.u[1] = ((const u32v4*)pl)[1];
    bAl1.u[0] = ((const u32v4*)(pl + 32))[0]; bAl1.u[1] = ((const u32v4*)(pl + 32))[1];
    ph = xh + (size_t)qB * DDIM + hf * 16;
    pl = xl + (size_t)qB * DDIM + hf * 16;
    bBh0.u[0] = ((const u32v4*)ph)[0];        bBh0.u[1] = ((const u32v4*)ph)[1];
    bBh1.u[0] = ((const u32v4*)(ph + 32))[0]; bBh1.u[1] = ((const u32v4*)(ph + 32))[1];
    bBl0.u[0] = ((const u32v4*)pl)[0];        bBl0.u[1] = ((const u32v4*)pl)[1];
    bBl1.u[0] = ((const u32v4*)(pl + 32))[0]; bBl1.u[1] = ((const u32v4*)(pl + 32))[1];
  }
  const float x2A = x2[qA];
  const float x2B = x2[qB];

  float worst0 = __builtin_inff(), worst1 = __builtin_inff();
  int   cnt0 = 0, cnt1 = 0;

  const int tilesPerHalf = (B2 / NT) / NSPLIT;   // 3125
  const int t0 = nhalf * tilesPerHalf;
  const int kbase = hf * 16;
  for (int t = t0 + wave; t < t0 + tilesPerHalf; t += NWAVES) {
    const size_t mrow = (size_t)(t * NT + qcol) * DDIM;
    const unsigned short* ahp = yh + mrow + kbase;
    const unsigned short* alp = yl + mrow + kbase;
    FragBF ah0, ah1, al0, al1;
    ah0.u[0] = ((const u32v4*)ahp)[0];        ah0.u[1] = ((const u32v4*)(ahp + 8))[0];
    ah1.u[0] = ((const u32v4*)(ahp + 32))[0]; ah1.u[1] = ((const u32v4*)(ahp + 40))[0];
    al0.u[0] = ((const u32v4*)alp)[0];        al0.u[1] = ((const u32v4*)(alp + 8))[0];
    al1.u[0] = ((const u32v4*)(alp + 32))[0]; al1.u[1] = ((const u32v4*)(alp + 40))[0];

    const f32v4* y2p = (const f32v4*)(y2 + t * NT + hf * 8);
    f32v4 y2a = y2p[0], y2b = y2p[1];

    v8f acc0 = {}, acc1 = {};
    // dot = xh*yh + xh*yl + xl*yh for both query tiles; A fragments reused 2x
    acc0 = __builtin_amdgcn_wmma_f32_16x16x32_bf16(false, ah0.v, false, bAh0.v, (short)0, acc0, false, false);
    acc0 = __builtin_amdgcn_wmma_f32_16x16x32_bf16(false, ah1.v, false, bAh1.v, (short)0, acc0, false, false);
    acc0 = __builtin_amdgcn_wmma_f32_16x16x32_bf16(false, ah0.v, false, bAl0.v, (short)0, acc0, false, false);
    acc0 = __builtin_amdgcn_wmma_f32_16x16x32_bf16(false, ah1.v, false, bAl1.v, (short)0, acc0, false, false);
    acc0 = __builtin_amdgcn_wmma_f32_16x16x32_bf16(false, al0.v, false, bAh0.v, (short)0, acc0, false, false);
    acc0 = __builtin_amdgcn_wmma_f32_16x16x32_bf16(false, al1.v, false, bAh1.v, (short)0, acc0, false, false);
    acc1 = __builtin_amdgcn_wmma_f32_16x16x32_bf16(false, ah0.v, false, bBh0.v, (short)0, acc1, false, false);
    acc1 = __builtin_amdgcn_wmma_f32_16x16x32_bf16(false, ah1.v, false, bBh1.v, (short)0, acc1, false, false);
    acc1 = __builtin_amdgcn_wmma_f32_16x16x32_bf16(false, ah0.v, false, bBl0.v, (short)0, acc1, false, false);
    acc1 = __builtin_amdgcn_wmma_f32_16x16x32_bf16(false, ah1.v, false, bBl1.v, (short)0, acc1, false, false);
    acc1 = __builtin_amdgcn_wmma_f32_16x16x32_bf16(false, al0.v, false, bBh0.v, (short)0, acc1, false, false);
    acc1 = __builtin_amdgcn_wmma_f32_16x16x32_bf16(false, al1.v, false, bBh1.v, (short)0, acc1, false, false);

    const unsigned int midx = (unsigned int)(t * NT + 8 * hf);
#pragma unroll
    for (int r = 0; r < 8; ++r) {
      float y2v = (r < 4) ? y2a[r] : y2b[r - 4];
      float d20 = fmaf(-2.0f, acc0[r], x2A + y2v);
      if (d20 < worst0) {            // clamp+pack only on the rare taken path
        q0[16 + cnt0] = ((u64)__float_as_uint(fmaxf(d20, 0.0f)) << 32) | (midx + r);
        ++cnt0;
      }
      float d21 = fmaf(-2.0f, acc1[r], x2B + y2v);
      if (d21 < worst1) {
        q1[16 + cnt1] = ((u64)__float_as_uint(fmaxf(d21, 0.0f)) << 32) | (midx + r);
        ++cnt1;
      }
    }
    if (__any(cnt0 > 8)) lane_compact(q0, worst0, cnt0);   // wave-uniform branch
    if (__any(cnt1 > 8)) lane_compact(q1, worst1, cnt1);
  }
  lane_compact(q0, worst0, cnt0);    // final flush: keep[0..15] sorted asc
  lane_compact(q1, worst1, cnt1);
  __syncthreads();

  // --- 16-way merge of sorted runs (8 waves x 2 lane-halves per query) ---
  if (threadIdx.x < QPB) {
    const int qq  = threadIdx.x;     // query within block
    const int qs  = qq >> 4;         // which lane-queue
    const int qc  = qq & 15;
    for (int reg = 0; reg < 16; ++reg) heads[qq][reg] = 0;
    u64* dst = cand + ((size_t)(qbase + qq) * NSPLIT + nhalf) * KTOP;
    for (int sel = 0; sel < KTOP; ++sel) {
      u64 best = ~0ull; int breg = 0;
      for (int reg = 0; reg < 16; ++reg) {
        // run: wave = reg>>1, half = reg&1; overrun reads hit UMAX pad
        u64 v = cbuf[reg >> 1][(reg & 1) * 16 + qc][qs * 32 + heads[qq][reg]];
        if (v < best) { best = v; breg = reg; }
      }
      heads[qq][breg]++;
      dst[sel] = best;
    }
  }
}

// ---------------------------------------------------------------------------
// Kernel C: merge the two database halves' sorted runs, unpack, final output.
// ---------------------------------------------------------------------------
__global__ void knn_merge2(const u64* __restrict__ cand,
                           float* __restrict__ out_ds, int* __restrict__ out_idx,
                           int B1) {
  int q = blockIdx.x * blockDim.x + threadIdx.x;
  if (q >= B1) return;
  const u64* r0 = cand + (size_t)q * (NSPLIT * KTOP);
  const u64* r1 = r0 + KTOP;
  int i0 = 0, i1 = 0;
  for (int s = 0; s < KTOP; ++s) {
    u64 a = r0[i0], b = r1[i1];
    u64 best;
    if (a < b) { best = a; ++i0; } else { best = b; ++i1; }
    out_ds [(size_t)q * KTOP + s] = sqrtf(__uint_as_float((unsigned int)(best >> 32)));
    out_idx[(size_t)q * KTOP + s] = (int)(unsigned int)(best & 0xffffffffu);
  }
}

// ---------------------------------------------------------------------------
extern "C" void kernel_launch(void* const* d_in, const int* in_sizes, int n_in,
                              void* d_out, int out_size, void* d_ws, size_t ws_size,
                              hipStream_t stream) {
  const float* x = (const float*)d_in[0];
  const float* y = (const float*)d_in[1];
  const int B1 = in_sizes[0] / DDIM;   // 4096
  const int B2 = in_sizes[1] / DDIM;   // 100000

  char* ws = (char*)d_ws;
  size_t off = 0;
  unsigned short* yh = (unsigned short*)(ws + off); off += (size_t)B2 * DDIM * 2;
  unsigned short* yl = (unsigned short*)(ws + off); off += (size_t)B2 * DDIM * 2;
  unsigned short* xh = (unsigned short*)(ws + off); off += (size_t)B1 * DDIM * 2;
  unsigned short* xl = (unsigned short*)(ws + off); off += (size_t)B1 * DDIM * 2;
  float* y2 = (float*)(ws + off); off += (size_t)B2 * sizeof(float);
  float* x2 = (float*)(ws + off); off += (size_t)B1 * sizeof(float);
  off = (off + 255) & ~(size_t)255;
  u64* cand = (u64*)(ws + off); off += (size_t)B1 * NSPLIT * KTOP * sizeof(u64);
  (void)ws_size; (void)n_in; (void)out_size;

  float* out_ds  = (float*)d_out;
  int*   out_idx = (int*)d_out + (size_t)B1 * KTOP;

  const int total = B1 + B2;
  knn_convert<<<(total + 255) / 256, 256, 0, stream>>>(x, y, xh, xl, yh, yl, x2, y2, B1, B2);
  knn_main<<<(B1 / QPB) * NSPLIT, 256, 0, stream>>>(xh, xl, yh, yl, x2, y2, cand, B1, B2);
  knn_merge2<<<(B1 + 255) / 256, 256, 0, stream>>>(cand, out_ds, out_idx, B1);
}
